// Struct_HeteroSAGE_5815385719107
// MI455X (gfx1250) — matrix operand order repaired and checked by reading, not compiled, not run on hardware
//
#include <hip/hip_runtime.h>
#include <hip/hip_bf16.h>

// ---------------------------------------------------------------------------
// Problem constants (match reference)
// ---------------------------------------------------------------------------
#define IN_DIM 768
#define HID 512
#define NS_N 100000
#define NW_N 100000
#define ND_N 1024
#define NP_N 1024
#define ESS_E 500000
#define ESW_E 500000
#define EF_E  200000
#define NEG_SLOPE 0.2f

typedef __attribute__((ext_vector_type(16))) _Float16 v16h;
typedef __attribute__((ext_vector_type(8)))  _Float16 v8h;
typedef __attribute__((ext_vector_type(8)))  float    v8f;

static inline int cdiv(int a, int b) { return (a + b - 1) / b; }

// ---------------------------------------------------------------------------
// GEMM: H[N x 512] = X[N x C] @ W[C x 512]  (+bias, optional relu)
// Block: 256 threads = 8 waves; block tile 64(M) x 128(N); K-step 32.
// A tile (64x32) staged once per block into LDS as f16 (row pad -> 40 f16 =
// 80 B, conflict-free b128 reads); all 8 waves build A fragments from LDS.
// B fragments loaded direct from global (W is small, lives in L2; each wave
// owns a distinct 16-column strip so there is no intra-block redundancy).
// WMMA: v_wmma_f32_16x16x32_f16, f32 accumulate.
// A frag (16-bit, 16x32): lane half selects K octet; m = lane&15.
// B frag (16-bit, 32x16): lane half selects K octet; n = lane&15.
// D layout: VGPR r -> (m = r + 8*(lane/16), n = lane&15).
// ---------------------------------------------------------------------------
__global__ __launch_bounds__(256)
void gemm_wmma_f16(const float* __restrict__ X, const float* __restrict__ W,
                   float* __restrict__ H, int N, int C,
                   const float* __restrict__ bias, int relu) {
  __shared__ _Float16 As[64 * 40];  // 64 rows x 32 k (f16), padded to 40

  const int tid  = threadIdx.x;
  const int lane = tid & 31;
  const int wave = tid >> 5;
  const int half = lane >> 4;
  const int mr   = lane & 15;
  const int n0   = blockIdx.y * 128 + wave * 16;
  const int row0 = blockIdx.x * 64;

  // cooperative staging coords: each thread loads 8 consecutive k of one row
  const int lrow = tid >> 2;        // 0..63
  const int lq   = (tid & 3) * 8;   // k offset: 0, 8, 16, 24
  const int grow = row0 + lrow;
  const bool in_range = (grow < N);
  const float* xrow = X + (size_t)grow * C;

  v8f acc[4] = {{}, {}, {}, {}};

  for (int k0 = 0; k0 < C; k0 += 32) {
    // ---- stage A tile into LDS (fp32 -> f16), zero-pad OOB rows ----
    __syncthreads();
    {
      float4 u0 = {0.f, 0.f, 0.f, 0.f}, u1 = {0.f, 0.f, 0.f, 0.f};
      if (in_range) {
        const float4* xp = (const float4*)(xrow + k0 + lq);
        u0 = xp[0];
        u1 = xp[1];
      }
      v8h s;
      s[0] = (_Float16)u0.x; s[1] = (_Float16)u0.y;
      s[2] = (_Float16)u0.z; s[3] = (_Float16)u0.w;
      s[4] = (_Float16)u1.x; s[5] = (_Float16)u1.y;
      s[6] = (_Float16)u1.z; s[7] = (_Float16)u1.w;
      *(v8h*)(As + lrow * 40 + lq) = s;
    }
    __syncthreads();

    // ---- B fragment: W[k0 + {8*half+i, 16+8*half+i}][n0 + mr] ----
    v16h b;
    {
      const float* wp = W + (size_t)(k0 + 8 * half) * HID + n0 + mr;
#pragma unroll
      for (int i = 0; i < 8; ++i) {
        b[i] = (_Float16)wp[(size_t)i * HID];
        b[8 + i] = (_Float16)wp[(size_t)(16 + i) * HID];
      }
    }

    // ---- 4 M-tiles: A fragments from LDS, branch-free WMMA ----
#pragma unroll
    for (int t = 0; t < 4; ++t) {
      const _Float16* ap = As + (t * 16 + mr) * 40 + 8 * half;
      v8h c0 = *(const v8h*)(ap);
      v8h c1 = *(const v8h*)(ap + 16);
      v16h a;
#pragma unroll
      for (int i = 0; i < 8; ++i) { a[i] = c0[i]; a[8 + i] = c1[i]; }
      acc[t] = __builtin_amdgcn_wmma_f32_16x16x32_f16(
          /*neg_a=*/false, a, /*neg_b=*/false, b,
          /*c_mod=*/(short)0, acc[t], /*reuse_a=*/false, /*reuse_b=*/false);
    }
  }

  const float bv = bias ? bias[n0 + mr] : 0.0f;
#pragma unroll
  for (int t = 0; t < 4; ++t) {
#pragma unroll
    for (int r = 0; r < 8; ++r) {
      const int row = row0 + t * 16 + 8 * half + r;
      if (row < N) {
        float v = acc[t][r] + bv;
        if (relu) v = fmaxf(v, 0.0f);
        H[(size_t)row * HID + n0 + mr] = v;
      }
    }
  }
}

// ---------------------------------------------------------------------------
// e[row] = dot(H[row, :512], a[:512]) ; one wave per row, 8 rows per block.
// ---------------------------------------------------------------------------
__global__ __launch_bounds__(256)
void rowdot512(const float* __restrict__ H, const float* __restrict__ a,
               float* __restrict__ e, int N) {
  const int lane = threadIdx.x & 31;
  const int wave = threadIdx.x >> 5;
  const int row  = blockIdx.x * 8 + wave;
  if (row >= N) return;
  const float4* hp = (const float4*)(H + (size_t)row * HID);
  const float4* ap = (const float4*)a;
  float s = 0.0f;
#pragma unroll
  for (int i = 0; i < 4; ++i) {
    float4 h = hp[lane + 32 * i];
    float4 w = ap[lane + 32 * i];
    s += h.x * w.x + h.y * w.y + h.z * w.z + h.w * w.w;
  }
#pragma unroll
  for (int m = 16; m; m >>= 1) s += __shfl_xor(s, m, 32);
  if (lane == 0) e[row] = s;
}

// ---------------------------------------------------------------------------
// Pass 1: e = leaky_relu(e_src[src] + e_dst[dst]); segment max into m[dst].
// Float atomic-max via signed/unsigned int-bit trick; m pre-set to NaN bytes.
// ---------------------------------------------------------------------------
__global__ __launch_bounds__(256)
void edge_max(const int* __restrict__ ei, int E,
              const float* __restrict__ es, const float* __restrict__ ed,
              float* __restrict__ ev, float* __restrict__ m) {
  const int e = blockIdx.x * 256 + threadIdx.x;
  if (e >= E) return;
  const int s = ei[e];
  const int d = ei[E + e];
  float v = es[s] + ed[d];
  v = (v > 0.0f) ? v : NEG_SLOPE * v;
  ev[e] = v;
  if (v >= 0.0f)
    atomicMax((int*)(m + d), __float_as_int(v));
  else
    atomicMin((unsigned int*)(m + d), __float_as_uint(v));
}

__global__ __launch_bounds__(256)
void clamp_nonfinite(float* __restrict__ m, int N) {
  const int i = blockIdx.x * 256 + threadIdx.x;
  if (i >= N) return;
  const float v = m[i];
  if (!__builtin_isfinite(v)) m[i] = 0.0f;
}

// ---------------------------------------------------------------------------
// Pass 2: ex = exp(e - m[dst]); denom[dst] += ex (segment sum).
// ---------------------------------------------------------------------------
__global__ __launch_bounds__(256)
void edge_exp(const int* __restrict__ ei, int E,
              const float* __restrict__ m, float* __restrict__ ev,
              float* __restrict__ denom) {
  const int e = blockIdx.x * 256 + threadIdx.x;
  if (e >= E) return;
  const int d = ei[E + e];
  const float x = __expf(ev[e] - m[d]);
  ev[e] = x;
  atomicAdd(denom + d, x);
}

// ---------------------------------------------------------------------------
// Pass 3: O[dst] += alpha * H[src]. 128 threads per edge, 4 channels each.
// 2 edges per 256-thread block; float4 gathers, float atomic scatters.
// ---------------------------------------------------------------------------
__global__ __launch_bounds__(256)
void edge_scatter(const int* __restrict__ ei, int E,
                  const float* __restrict__ ev, const float* __restrict__ denom,
                  const float* __restrict__ Hs, float* __restrict__ O) {
  const int e = blockIdx.x * 2 + (threadIdx.x >> 7);
  if (e >= E) return;
  const int c = (threadIdx.x & 127) * 4;
  const int s = ei[e];
  const int d = ei[E + e];
  const float alpha = ev[e] / (denom[d] + 1e-16f);
  const float4 h = *(const float4*)(Hs + (size_t)s * HID + c);
  float* o = O + (size_t)d * HID + c;
  atomicAdd(o + 0, alpha * h.x);
  atomicAdd(o + 1, alpha * h.y);
  atomicAdd(o + 2, alpha * h.z);
  atomicAdd(o + 3, alpha * h.w);
}

// ---------------------------------------------------------------------------
// O = relu_opt( (O + b0 + b1 + b2) * scale )   (b1/b2 may be null)
// ---------------------------------------------------------------------------
__global__ __launch_bounds__(256)
void finalize_node(float* __restrict__ O,
                   const float* __restrict__ b0, const float* __restrict__ b1,
                   const float* __restrict__ b2, float scale, int N, int relu) {
  const long long total = (long long)N * HID;
  const long long i = (long long)blockIdx.x * 256 + threadIdx.x;
  if (i >= total) return;
  const int c = (int)(i & (HID - 1));
  float bb = 0.0f;
  if (b0) bb += b0[c];
  if (b1) bb += b1[c];
  if (b2) bb += b2[c];
  float v = (O[i] + bb) * scale;
  if (relu) v = fmaxf(v, 0.0f);
  O[i] = v;
}

// ---------------------------------------------------------------------------
// Host orchestration
// ---------------------------------------------------------------------------
extern "C" void kernel_launch(void* const* d_in, const int* in_sizes, int n_in,
                              void* d_out, int out_size, void* d_ws, size_t ws_size,
                              hipStream_t stream) {
  (void)in_sizes; (void)n_in; (void)out_size; (void)ws_size;

  const float* x0[4] = {(const float*)d_in[0], (const float*)d_in[1],
                        (const float*)d_in[2], (const float*)d_in[3]};
  const int* ei[6]   = {(const int*)d_in[4], (const int*)d_in[5],
                        (const int*)d_in[6], (const int*)d_in[7],
                        (const int*)d_in[8], (const int*)d_in[9]};
  const float* Wl[3]  = {(const float*)d_in[10], (const float*)d_in[14],
                         (const float*)d_in[18]};
  const float* asl[3] = {(const float*)d_in[11], (const float*)d_in[15],
                         (const float*)d_in[19]};
  const float* adl[3] = {(const float*)d_in[12], (const float*)d_in[16],
                         (const float*)d_in[20]};
  const float* bl[3]  = {(const float*)d_in[13], (const float*)d_in[17],
                         (const float*)d_in[21]};
  const float* fcw = (const float*)d_in[22];
  const float* fcb = (const float*)d_in[23];

  const int Ntype[4] = {NS_N, NW_N, ND_N, NP_N};
  const int srcT[6]  = {0, 0, 1, 2, 1, 3};   // relation source node type
  const int dstT[6]  = {0, 1, 2, 1, 3, 1};   // relation dest node type
  const int Ecnt[6]  = {ESS_E, ESW_E, EF_E, EF_E, EF_E, EF_E};

  // ---- carve workspace ----
  float* ws = (float*)d_ws;
  size_t off = 0;
  auto carve = [&](size_t n) { float* p = ws + off; off += n; return p; };
  float* bufA = carve((size_t)NS_N * HID);
  float* bufB = carve((size_t)NS_N * HID);
  float* setA[4]; float* setB[4];
  for (int t = 0; t < 4; ++t) setA[t] = carve((size_t)Ntype[t] * HID);
  for (int t = 0; t < 4; ++t) setB[t] = carve((size_t)Ntype[t] * HID);
  float* eS    = carve(NS_N);
  float* eD    = carve(NS_N);
  float* mbuf  = carve(NS_N);
  float* dbuf  = carve(NS_N);
  float* evbuf = carve(ESS_E);

  auto gemm = [&](const float* X, const float* W, float* H, int N, int C,
                  const float* bias, int relu) {
    dim3 grid(cdiv(N, 64), HID / 128);
    gemm_wmma_f16<<<grid, 256, 0, stream>>>(X, W, H, N, C, bias, relu);
  };

  const float* xin[4] = {x0[0], x0[1], x0[2], x0[3]};
  int cin = IN_DIM;

  for (int l = 0; l < 3; ++l) {
    float** oset = (l & 1) ? setB : setA;
    for (int t = 0; t < 4; ++t)
      hipMemsetAsync(oset[t], 0, (size_t)Ntype[t] * HID * sizeof(float), stream);

    for (int r = 0; r < 6; ++r) {
      const int st = srcT[r], dt = dstT[r];
      const int Nsrc = Ntype[st], Ndst = Ntype[dt];
      const int E = Ecnt[r];
      const float* Wr = Wl[l] + (size_t)r * cin * HID;

      gemm(xin[st], Wr, bufA, Nsrc, cin, nullptr, 0);
      const float* hdst = bufA;
      if (st != dt) { gemm(xin[dt], Wr, bufB, Ndst, cin, nullptr, 0); hdst = bufB; }

      rowdot512<<<cdiv(Nsrc, 8), 256, 0, stream>>>(bufA, asl[l] + (size_t)r * HID, eS, Nsrc);
      rowdot512<<<cdiv(Ndst, 8), 256, 0, stream>>>(hdst, adl[l] + (size_t)r * HID, eD, Ndst);

      hipMemsetAsync(mbuf, 0xFF, (size_t)Ndst * sizeof(float), stream);  // NaN sentinel
      hipMemsetAsync(dbuf, 0,    (size_t)Ndst * sizeof(float), stream);

      edge_max<<<cdiv(E, 256), 256, 0, stream>>>(ei[r], E, eS, eD, evbuf, mbuf);
      clamp_nonfinite<<<cdiv(Ndst, 256), 256, 0, stream>>>(mbuf, Ndst);
      edge_exp<<<cdiv(E, 256), 256, 0, stream>>>(ei[r], E, mbuf, evbuf, dbuf);
      edge_scatter<<<cdiv(E, 2), 256, 0, stream>>>(ei[r], E, evbuf, dbuf, bufA, oset[dt]);
    }

    // bias + HeteroConv mean (w collects 3 relations: 1, 3, 5) + relu
    const float* b = bl[l];
    finalize_node<<<cdiv(NS_N * (HID / 256), 1), 256, 0, stream>>>(
        oset[0], b + 0 * HID, nullptr, nullptr, 1.0f, NS_N, 1);
    finalize_node<<<cdiv(NW_N * (HID / 256), 1), 256, 0, stream>>>(
        oset[1], b + 1 * HID, b + 3 * HID, b + 5 * HID, 1.0f / 3.0f, NW_N, 1);
    finalize_node<<<cdiv(ND_N * (HID / 256), 1), 256, 0, stream>>>(
        oset[2], b + 2 * HID, nullptr, nullptr, 1.0f, ND_N, 1);
    finalize_node<<<cdiv(NP_N * (HID / 256), 1), 256, 0, stream>>>(
        oset[3], b + 4 * HID, nullptr, nullptr, 1.0f, NP_N, 1);

    for (int t = 0; t < 4; ++t) xin[t] = oset[t];
    cin = HID;
  }

  // final FC on word nodes: d_out = xw @ fc_w + fc_b
  gemm(xin[1], fcw, (float*)d_out, NW_N, HID, fcb, 0);
}
